// SparseSpikingConv2D_35442070126944
// MI455X (gfx1250) — matrix operand order repaired and checked by reading, not compiled, not run on hardware
//
#include <hip/hip_runtime.h>
#include <hip/hip_bf16.h>

typedef unsigned short u16;
typedef unsigned int   u32;

typedef __attribute__((ext_vector_type(16))) __bf16 v16bf;
typedef __attribute__((ext_vector_type(8)))  float  v8f;
typedef int i32x4 __attribute__((vector_size(16)));

#define AS1 __attribute__((address_space(1)))
#define AS3 __attribute__((address_space(3)))

struct U32x8 { uint4 lo; uint4 hi; };

#define CIN   64
#define COUT  128
#define HW    64
#define NB    32
#define KTAPS 9
#define KSTEPS 18      // 9 taps * 2 halves of Cin (32 each)
#define TH 8           // h rows per block tile (2 per wave -> B reuse x2)
#define TW 16          // w cols per block tile
#define PR (TH + 2)    // patch rows  (halo)
#define PC (TW + 2)    // patch cols  (halo)

// ---- workspace layout (bytes) ----
#define XT_BYTES   (NB * HW * HW * CIN * 2)            // 16,777,216  bf16 x, [n][h][w][cin]
#define BSW_BYTES  (KSTEPS * 8 * 32 * 8 * 4)           // 147,456     swizzled B fragments
#define BSW_OFF    XT_BYTES
#define NRM_OFF    (XT_BYTES + BSW_BYTES)

#if __has_builtin(__builtin_amdgcn_global_load_async_to_lds_b128)
#define ASYNC_LDS 1
#else
#define ASYNC_LDS 0
#endif

__device__ __forceinline__ u32 f32_to_bf16_rne(float f) {
    u32 u = __float_as_uint(f);
    return (u + 0x7FFFu + ((u >> 16) & 1u)) >> 16;
}

// Kernel 1: x [n][cin][h][w] f32  ->  xt [n][h][w][cin] bf16
__global__ __launch_bounds__(256) void k_xpose_bf16(
        const float* __restrict__ x, u16* __restrict__ xt) {
    int tid = blockIdx.x * 256 + threadIdx.x;     // 8,388,608 elements
    int w   =  tid        & 63;
    int h   = (tid >>  6) & 63;
    int cin = (tid >> 12) & 63;
    int n   =  tid >> 18;
    u32 r = f32_to_bf16_rne(x[tid]);
    xt[((((n * HW) + h) * HW + w) << 6) + cin] = (u16)r;
}

// Kernel 2: kernel [576][128] f32 -> WMMA B-fragment swizzled bf16 pairs.
// dword index = ((s*8 + ntile)*32 + lane)*8 + j
// B 32x16 layout: lane<16 -> K {0..7,16..23}, lane>=16 -> K {8..15,24..31}; VGPR j holds a consecutive K pair.
__global__ __launch_bounds__(256) void k_swizzle_b(
        const float* __restrict__ kern, u32* __restrict__ bsw) {
    int tid  = blockIdx.x * 256 + threadIdx.x;    // 36,864 dwords
    int j    =  tid        & 7;
    int lane = (tid >>  3) & 31;
    int t    = (tid >>  8) & 7;
    int s    =  tid >> 11;
    int krow = s * 32 + ((lane >> 4) << 3) + 2 * j + ((j >> 2) << 3);
    int co   = t * 16 + (lane & 15);
    u32 u0 = f32_to_bf16_rne(kern[ krow      * COUT + co]);
    u32 u1 = f32_to_bf16_rne(kern[(krow + 1) * COUT + co]);
    bsw[tid] = (u0 & 0xFFFFu) | (u1 << 16);
}

// Kernel 3: norm[cout] = sum over 576 of kernel^2
__global__ void k_norm(const float* __restrict__ kern, float* __restrict__ nrm) {
    int c = threadIdx.x;                          // 128 threads
    float s = 0.0f;
    for (int i = 0; i < KTAPS * CIN; ++i) {
        float v = kern[i * COUT + c];
        s += v * v;
    }
    nrm[c] = s;
}

// Kernel 4: implicit-GEMM conv (WMMA bf16) + fused LIF epilogue.
// Block tile: 8 h-rows x 16 w-cols x 128 couts. 8 waves:
//   (wave&3)*2   -> pair of h rows (two 16-pixel M tiles -> each B frag feeds 2 WMMAs)
//   wave>>2      -> cout group (0/1), 4 x 16-cout N tiles each
__global__ __launch_bounds__(256) void k_conv_lif(
        const uint4* __restrict__ xt4,      // bf16 [n][h][w][cin], as uint4 (8 cin / vec)
        const uint4* __restrict__ bsw4,     // swizzled B fragments
        const float* __restrict__ mem,
        const float* __restrict__ nrm,
        const float* __restrict__ beta_p,
        const float* __restrict__ bvec,
        float* __restrict__ out_spk,
        float* __restrict__ out_mem) {
    __shared__ uint4 patch4[PR * PC * 8];   // [row][col][cin/8]  (23,040 B)

    const int tid  = threadIdx.x;
    const int lane = tid & 31;
    const int wave = tid >> 5;
    const int n  = blockIdx.z;
    const int h0 = blockIdx.y * TH;
    const int w0 = blockIdx.x * TW;

    // ---- stage input halo patch: 10 x 18 x 64 bf16, b128 granules ----
    // In-bounds granules go straight to LDS via async loads (ASYNCcnt path);
    // halo out-of-bounds granules are zero-filled with DS stores (disjoint
    // addresses, so no ordering hazard against the async writes).
    const uint4 zero4 = make_uint4(0u, 0u, 0u, 0u);
    for (int cid = tid; cid < PR * PC * 8; cid += 256) {
        int row = cid / (PC * 8);
        int rem = cid - row * (PC * 8);
        int col = rem >> 3;
        int seg = rem & 7;
        int gh = h0 - 1 + row;
        int gw = w0 - 1 + col;
        bool inb = ((unsigned)gh < (unsigned)HW) && ((unsigned)gw < (unsigned)HW);
#if ASYNC_LDS
        if (inb) {
            __builtin_amdgcn_global_load_async_to_lds_b128(
                (AS1 i32x4*)(xt4 + ((((n * HW) + gh) * HW + gw) * 8 + seg)),
                (AS3 i32x4*)(patch4 + cid),
                0, 0);
        } else {
            patch4[cid] = zero4;
        }
#else
        patch4[cid] = inb ? xt4[(((n * HW) + gh) * HW + gw) * 8 + seg] : zero4;
#endif
    }
#if ASYNC_LDS
#if __has_builtin(__builtin_amdgcn_s_wait_asynccnt)
    __builtin_amdgcn_s_wait_asynccnt(0);
#else
    asm volatile("s_wait_asynccnt 0x0" ::: "memory");
#endif
#endif
    __syncthreads();

    const int mt2 = (wave & 3) << 1;  // first of the wave's two h rows
    const int ng  = wave >> 2;        // cout group
    const int mrow   = lane & 15;     // A row == w col within tile
    const int hiHalf = (lane >> 4) & 1;

    v8f acc[2][4] = {};

    for (int s = 0; s < KSTEPS; ++s) {
        int kk   = s >> 1;
        int dy   = kk / 3;
        int dx   = kk - dy * 3;
        int cin0 = (s & 1) << 5;
        // A fragment: lane<16 holds K {0..7,16..23}, lane>=16 holds K {8..15,24..31}
        int cb = cin0 + (hiHalf << 3);
        const uint4* ap0 = patch4 + ((mt2 + dy) * PC + (mrow + dx)) * 8 + (cb >> 3);
        U32x8 au0, au1;
        au0.lo = ap0[0];
        au0.hi = ap0[2];               // +16 bf16 elements
        au1.lo = ap0[PC * 8];          // next h row
        au1.hi = ap0[PC * 8 + 2];
        v16bf a0 = __builtin_bit_cast(v16bf, au0);
        v16bf a1 = __builtin_bit_cast(v16bf, au1);
#pragma unroll
        for (int t = 0; t < 4; ++t) {
            int nt = ng * 4 + t;
            const uint4* bp = bsw4 + (((s * 8 + nt) * 32) + lane) * 2;
            U32x8 bu;
            bu.lo = bp[0];
            bu.hi = bp[1];
            v16bf bf = __builtin_bit_cast(v16bf, bu);
            acc[0][t] = __builtin_amdgcn_wmma_f32_16x16x32_bf16(
                false, a0, false, bf, (short)0, acc[0][t], false, false);
            acc[1][t] = __builtin_amdgcn_wmma_f32_16x16x32_bf16(
                false, a1, false, bf, (short)0, acc[1][t], false, false);
        }
    }

    // ---- fused LIF epilogue ----
    // C layout: VGPR r, lane -> M = r + 8*(lane>=16), N = lane%16
    const float beta = beta_p[0];
    const float omb  = 1.0f - beta;
    const int   ncol = lane & 15;
    const int   mhi  = hiHalf << 3;
    const int   hb   = h0 + mt2;
#pragma unroll
    for (int t = 0; t < 4; ++t) {
        int co = ng * 64 + t * 16 + ncol;
        float rn = nrm[co] + 1e-8f;
        float bb = bvec[co];
#pragma unroll
        for (int m = 0; m < 2; ++m) {
            int h = hb + m;
#pragma unroll
            for (int r = 0; r < 8; ++r) {
                int w = w0 + r + mhi;
                int idx = (((n * COUT) + co) * HW + h) * HW + w;
                float conv = acc[m][t][r];
                float nm   = mem[idx] * beta + conv * omb;
                float mthr = nm / rn - bb;
                float spk  = (mthr > 0.0f ? 1.0f : 0.0f)
                           + (mthr > 1.0f ? 1.0f : 0.0f)
                           + (mthr > 2.0f ? 1.0f : 0.0f)
                           + (mthr > 3.0f ? 1.0f : 0.0f);
                out_spk[idx] = spk;
                out_mem[idx] = (spk > 0.0f) ? 0.0f : nm;
            }
        }
    }
}

extern "C" void kernel_launch(void* const* d_in, const int* in_sizes, int n_in,
                              void* d_out, int out_size, void* d_ws, size_t ws_size,
                              hipStream_t stream) {
    const float* x    = (const float*)d_in[0];
    const float* mem  = (const float*)d_in[1];
    const float* kern = (const float*)d_in[2];
    const float* beta = (const float*)d_in[3];
    const float* bvec = (const float*)d_in[4];

    u16*   xt  = (u16*)d_ws;
    u32*   bsw = (u32*)((char*)d_ws + BSW_OFF);
    float* nrm = (float*)((char*)d_ws + NRM_OFF);

    float* out_spk = (float*)d_out;
    float* out_mem = out_spk + (size_t)NB * COUT * HW * HW;

    // 1) x -> bf16 [n][h][w][cin]
    k_xpose_bf16<<<dim3((NB * CIN * HW * HW) / 256), dim3(256), 0, stream>>>(x, xt);
    // 2) weights -> WMMA B-fragment layout
    k_swizzle_b<<<dim3((KSTEPS * 8 * 32 * 8) / 256), dim3(256), 0, stream>>>(kern, bsw);
    // 3) per-cout kernel norm
    k_norm<<<dim3(1), dim3(COUT), 0, stream>>>(kern, nrm);
    // 4) WMMA conv + fused LIF
    k_conv_lif<<<dim3(HW / TW, HW / TH, NB), dim3(256), 0, stream>>>(
        (const uint4*)xt, (const uint4*)bsw, mem, nrm, beta, bvec, out_spk, out_mem);
}
